// PL1InverseApprox_66159676228022
// MI455X (gfx1250) — matrix operand order repaired and checked by reading, not compiled, not run on hardware
//
#include <hip/hip_runtime.h>
#include <hip/hip_bf16.h>

typedef __attribute__((ext_vector_type(16))) _Float16 v16h;
typedef __attribute__((ext_vector_type(8)))  float    v8f;

#define NEG_SLOPE 0.2f
#define D_IN  128
#define D_OUT 64

// ---------------------------------------------------------------------------
// Kernel 0: init out=0, m=-inf, denom=0
// ---------------------------------------------------------------------------
__global__ void gat_init(float* __restrict__ out, float* __restrict__ m,
                         float* __restrict__ denom, int N) {
    long long t = (long long)blockIdx.x * blockDim.x + threadIdx.x;
    long long total = (long long)N * D_OUT;
    if (t < total) out[t] = 0.0f;
    if (t < N) { m[t] = -__builtin_inff(); denom[t] = 0.0f; }
}

// ---------------------------------------------------------------------------
// Kernel 1: h = x @ W via v_wmma_f32_16x16x32_f16
// 256 threads = 8 waves; each wave computes a 16-row x 64-col strip of h.
// W (128x64 f32) is staged once per block into LDS, PRE-SWIZZLED into the
// exact WMMA B-fragment layout so each fragment is one aligned v16h LDS load.
//
// Fragment (kc, ct) of B (32x16 f16): lane L holds col = ct*16 + (L&15),
// K = kc*32 + (L<16 ? 0 : 8) + j + (j>=8 ? 8 : 0), j = 0..15 (halves packed 2 per VGPR).
// LDS layout: sFrag[((kc*4+ct)*32 + lane)*16 + j]
// ---------------------------------------------------------------------------
__global__ __launch_bounds__(256)
void gat_gemm_wmma(const float* __restrict__ x, const float* __restrict__ W,
                   float* __restrict__ h, int N) {
    __shared__ _Float16 sFrag[16 * 32 * 16];   // 16 frags * 32 lanes * 16 halves = 16 KB

    const int t = threadIdx.x;
    // stage W -> LDS in fragment layout: 512 fragment-rows of 16 halves,
    // each thread builds 2 rows (16 strided W reads + one 32B LDS store each)
    for (int r = t; r < 512; r += 256) {
        const int frag = r >> 5;          // 0..15  (kc*4 + ct)
        const int lane = r & 31;
        const int kc   = frag >> 2;
        const int ct   = frag & 3;
        const int col  = ct * 16 + (lane & 15);
        const int k0   = kc * 32 + ((lane & 16) ? 8 : 0);
        v16h tmp;
#pragma unroll
        for (int j = 0; j < 16; ++j)
            tmp[j] = (_Float16)W[(k0 + j + (j >= 8 ? 8 : 0)) * D_OUT + col];
        *(v16h*)(sFrag + r * 16) = tmp;
    }
    __syncthreads();

    const int wave = t >> 5;
    const int lane = t & 31;
    const int rowTile = blockIdx.x * 8 + wave;
    if (rowTile * 16 >= N) return;            // whole-wave uniform exit
    const bool fullTile = (rowTile * 16 + 16 <= N);   // wave-uniform

    // ---- A fragments: 16x32 f16 per K-chunk, per ISA layout ----
    // lanes 0-15 hold M=lane, K = {0..7, 16..23}; lanes 16-31 hold M=lane-16,
    // K = {8..15, 24..31}  (element i of v16h = half i, VGPRv = elems 2v,2v+1)
    int row = rowTile * 16 + (lane & 15);
    if (row >= N) row = N - 1;                // clamp keeps EXEC all-ones
    const int kbaseA = (lane < 16) ? 0 : 8;
    const float* xrow = x + (size_t)row * D_IN;

    v16h A[4];
#pragma unroll
    for (int kc = 0; kc < 4; ++kc) {
        const float* p = xrow + kc * 32 + kbaseA;
        float4 f0 = *(const float4*)(p);
        float4 f1 = *(const float4*)(p + 4);
        float4 f2 = *(const float4*)(p + 16);
        float4 f3 = *(const float4*)(p + 20);
        A[kc][0]  = (_Float16)f0.x; A[kc][1]  = (_Float16)f0.y;
        A[kc][2]  = (_Float16)f0.z; A[kc][3]  = (_Float16)f0.w;
        A[kc][4]  = (_Float16)f1.x; A[kc][5]  = (_Float16)f1.y;
        A[kc][6]  = (_Float16)f1.z; A[kc][7]  = (_Float16)f1.w;
        A[kc][8]  = (_Float16)f2.x; A[kc][9]  = (_Float16)f2.y;
        A[kc][10] = (_Float16)f2.z; A[kc][11] = (_Float16)f2.w;
        A[kc][12] = (_Float16)f3.x; A[kc][13] = (_Float16)f3.y;
        A[kc][14] = (_Float16)f3.z; A[kc][15] = (_Float16)f3.w;
    }

#pragma unroll
    for (int ct = 0; ct < 4; ++ct) {
        v8f acc = {};
#pragma unroll
        for (int kc = 0; kc < 4; ++kc) {
            // one aligned 32B vector load = the whole B fragment for this lane
            v16h B = *(const v16h*)(sFrag + (((kc * 4 + ct) * 32) + lane) * 16);
            acc = __builtin_amdgcn_wmma_f32_16x16x32_f16(
                false, A[kc], false, B, (short)0, acc, false, false);
        }
        // ---- store C: VGPR r holds M = r (lanes 0-15) / r+8 (lanes 16-31) ----
        const int col   = ct * 16 + (lane & 15);
        const int rbase = rowTile * 16 + ((lane < 16) ? 0 : 8);
        float* hp = h + (size_t)rbase * D_OUT + col;
        if (fullTile) {
            // unguarded: 8 stores off one base pointer, offsets fold into the
            // global_store_b32 immediate (no per-store exec/branch overhead)
#pragma unroll
            for (int r = 0; r < 8; ++r)
                hp[(size_t)r * D_OUT] = acc[r];
        } else {
#pragma unroll
            for (int r = 0; r < 8; ++r)
                if (rbase + r < N) hp[(size_t)r * D_OUT] = acc[r];
        }
    }
}

// ---------------------------------------------------------------------------
// Kernel 2: per-node attention dots a_src[n] = h[n].att_src, a_dst likewise
// ---------------------------------------------------------------------------
__global__ void gat_att(const float* __restrict__ h,
                        const float* __restrict__ att_src,
                        const float* __restrict__ att_dst,
                        float* __restrict__ as_n, float* __restrict__ ad_n, int N) {
    int n = blockIdx.x * blockDim.x + threadIdx.x;
    if (n >= N) return;
    const float* hr = h + (size_t)n * D_OUT;
    float s = 0.0f, d = 0.0f;
#pragma unroll
    for (int c = 0; c < D_OUT; c += 4) {
        float4 v = *(const float4*)(hr + c);
        s += v.x * att_src[c] + v.y * att_src[c + 1] + v.z * att_src[c + 2] + v.w * att_src[c + 3];
        d += v.x * att_dst[c] + v.y * att_dst[c + 1] + v.z * att_dst[c + 2] + v.w * att_dst[c + 3];
    }
    as_n[n] = s; ad_n[n] = d;
}

// IEEE-exact float atomic max via int/uint atomics (valid with -inf init)
__device__ __forceinline__ void atomicMaxF(float* addr, float v) {
    if (v >= 0.0f)
        atomicMax((int*)addr, __float_as_int(v));
    else
        atomicMin((unsigned int*)addr, __float_as_uint(v));
}

// ---------------------------------------------------------------------------
// Kernel 3: e = leaky_relu(a_src[src]+a_dst[dst]); segment max into m[dst]
// Edges 0..E-1 from edge_index, edges E..E+N-1 are self loops.
// ---------------------------------------------------------------------------
__global__ void gat_edge_max(const int* __restrict__ src, const int* __restrict__ dst,
                             const float* __restrict__ as_n, const float* __restrict__ ad_n,
                             float* __restrict__ e_buf, float* __restrict__ m,
                             int E, int N) {
    int t = blockIdx.x * blockDim.x + threadIdx.x;
    if (t >= E + N) return;
    int s, d;
    if (t < E) { s = src[t]; d = dst[t]; } else { s = t - E; d = s; }
    float e = as_n[s] + ad_n[d];
    e = (e > 0.0f) ? e : NEG_SLOPE * e;
    e_buf[t] = e;
    atomicMaxF(&m[d], e);
}

// ---------------------------------------------------------------------------
// Kernel 4: ex = exp(e - m[dst]); segment sum into denom[dst]
// ---------------------------------------------------------------------------
__global__ void gat_edge_exp(const int* __restrict__ dst,
                             float* __restrict__ e_buf,
                             const float* __restrict__ m, float* __restrict__ denom,
                             int E, int N) {
    int t = blockIdx.x * blockDim.x + threadIdx.x;
    if (t >= E + N) return;
    int d = (t < E) ? dst[t] : (t - E);
    float v = expf(e_buf[t] - m[d]);
    e_buf[t] = v;
    atomicAdd(&denom[d], v);
}

// ---------------------------------------------------------------------------
// Kernel 5: out[dst] += alpha * h[src].  One wave32 per edge, lane = col pair.
// ---------------------------------------------------------------------------
__global__ __launch_bounds__(256)
void gat_scatter(const int* __restrict__ src, const int* __restrict__ dst,
                 const float* __restrict__ h, const float* __restrict__ e_buf,
                 const float* __restrict__ denom, float* __restrict__ out,
                 int E, int N) {
    long long gtid = (long long)blockIdx.x * blockDim.x + threadIdx.x;
    long long k = gtid >> 5;               // one wave per edge
    int lane = threadIdx.x & 31;
    if (k >= (long long)E + N) return;
    int s, d;
    if (k < E) { s = src[k]; d = dst[k]; } else { s = (int)(k - E); d = s; }
    float alpha = e_buf[k] / denom[d];
    float2 hv = *(const float2*)(h + (size_t)s * D_OUT + lane * 2);
    float* op = out + (size_t)d * D_OUT + lane * 2;
    atomicAdd(op,     hv.x * alpha);
    atomicAdd(op + 1, hv.y * alpha);
}

// ---------------------------------------------------------------------------
// Kernel 6: out = elu(out + bias)
// ---------------------------------------------------------------------------
__global__ void gat_finish(float* __restrict__ out, const float* __restrict__ bias, int N) {
    long long t = (long long)blockIdx.x * blockDim.x + threadIdx.x;
    long long total = (long long)N * D_OUT;
    if (t >= total) return;
    float v = out[t] + bias[t & (D_OUT - 1)];
    out[t] = (v > 0.0f) ? v : (expf(v) - 1.0f);
}

// ---------------------------------------------------------------------------
extern "C" void kernel_launch(void* const* d_in, const int* in_sizes, int n_in,
                              void* d_out, int out_size, void* d_ws, size_t ws_size,
                              hipStream_t stream) {
    const float* x       = (const float*)d_in[0];   // [N, 128]
    const float* W       = (const float*)d_in[1];   // [128, 64]
    const float* att_src = (const float*)d_in[2];   // [64]
    const float* att_dst = (const float*)d_in[3];   // [64]
    const float* bias    = (const float*)d_in[4];   // [64]
    const int*   ei      = (const int*)d_in[5];     // [2, E] flat

    const int N = in_sizes[0] / D_IN;
    const int E = in_sizes[5] / 2;
    const int* src = ei;
    const int* dst = ei + E;

    float* out = (float*)d_out;

    // workspace layout (floats)
    float* ws    = (float*)d_ws;
    size_t off   = 0;
    float* h     = ws + off; off += (size_t)N * D_OUT;
    float* as_n  = ws + off; off += (size_t)N;
    float* ad_n  = ws + off; off += (size_t)N;
    float* m     = ws + off; off += (size_t)N;
    float* denom = ws + off; off += (size_t)N;
    float* e_buf = ws + off; off += (size_t)E + N;
    (void)ws_size; (void)n_in; (void)out_size;

    const int T = E + N;

    // 0) init
    {
        long long total = (long long)N * D_OUT;
        int blocks = (int)((total + 255) / 256);
        gat_init<<<blocks, 256, 0, stream>>>(out, m, denom, N);
    }
    // 1) WMMA GEMM: h = x @ W
    {
        int rowTiles = (N + 15) / 16;
        int blocks = (rowTiles + 7) / 8;
        gat_gemm_wmma<<<blocks, 256, 0, stream>>>(x, W, h, N);
    }
    // 2) attention dots
    gat_att<<<(N + 255) / 256, 256, 0, stream>>>(h, att_src, att_dst, as_n, ad_n, N);
    // 3) leaky-relu + segment max
    gat_edge_max<<<(T + 255) / 256, 256, 0, stream>>>(src, dst, as_n, ad_n, e_buf, m, E, N);
    // 4) exp + segment sum
    gat_edge_exp<<<(T + 255) / 256, 256, 0, stream>>>(dst, e_buf, m, denom, E, N);
    // 5) weighted scatter-add (one wave per edge)
    {
        long long threads = (long long)T * 32;
        int blocks = (int)((threads + 255) / 256);
        gat_scatter<<<blocks, 256, 0, stream>>>(src, dst, h, e_buf, denom, out, E, N);
    }
    // 6) bias + ELU
    {
        long long total = (long long)N * D_OUT;
        int blocks = (int)((total + 255) / 256);
        gat_finish<<<blocks, 256, 0, stream>>>(out, bias, N);
    }
}
